// RelativePosMultiHeadAttention_55293408969233
// MI455X (gfx1250) — matrix operand order repaired and checked by reading, not compiled, not run on hardware
//
#include <hip/hip_runtime.h>
#include <hip/hip_bf16.h>
#include <cstdint>
#include <cstddef>

// ---------------------------------------------------------------------------
// RelativePosMultiHeadAttention for MI455X (gfx1250), wave32 + WMMA bf16.
// All fragments are loaded as pairs of 128-bit ops (fragment-ready layouts).
// ---------------------------------------------------------------------------

typedef __bf16 bf16_t;
typedef __attribute__((ext_vector_type(16))) __bf16 v16bf;
typedef __attribute__((ext_vector_type(8)))  float  v8f;

namespace {

constexpr int kB   = 8;
constexpr int kS   = 1024;
constexpr int kDIM = 512;
constexpr int kNH  = 8;
constexpr int kHD  = 64;
constexpr float kInvSqrtDim = 0.04419417382415922f;  // 1/sqrt(512)

__device__ __forceinline__ unsigned short f2bf_bits(float f) {
  unsigned u = __builtin_bit_cast(unsigned, f);
  unsigned r = u + 0x7FFFu + ((u >> 16) & 1u);   // round-to-nearest-even
  return (unsigned short)(r >> 16);
}
__device__ __forceinline__ bf16_t f2bf(float f) {
  unsigned short h = f2bf_bits(f);
  return __builtin_bit_cast(bf16_t, h);
}
__device__ __forceinline__ unsigned pack2(float a, float b) {
  return (unsigned)f2bf_bits(a) | ((unsigned)f2bf_bits(b) << 16);
}
__device__ __forceinline__ float bf2f(bf16_t b) { return (float)b; }

__device__ __forceinline__ v8f zero8() {
  v8f z;
#pragma unroll
  for (int i = 0; i < 8; ++i) z[i] = 0.f;
  return z;
}

// A 16x16x32 bf16 fragment is 16 lanes-private bf16 = 32 bytes = 2 x b128.
union Frag {
  v16bf v;
  uint4 q[2];
};
__device__ __forceinline__ Frag load_frag(const bf16_t* p0, const bf16_t* p1) {
  Frag f;
  f.q[0] = *(const uint4*)p0;
  f.q[1] = *(const uint4*)p1;
  return f;
}
struct B8 { bf16_t e[8]; };
__device__ __forceinline__ B8 load8(const bf16_t* p) {
  union { uint4 q; B8 b; } u;
  u.q = *(const uint4*)p;
  return u.b;
}

__device__ __forceinline__ v8f wmma_bf16(v16bf a, v16bf b, v8f c) {
  return __builtin_amdgcn_wmma_f32_16x16x32_bf16(false, a, false, b,
                                                 (short)0, c, false, false);
}

}  // namespace

// ---------------------------------------------------------------------------
// Kernel 0: W [o,i] f32 -> WF bf16 in B-fragment order:
//   WF[((kc*(DIM/16) + nt)*32 + lane)*16 + j] = W[nt*16 + (lane&15)]
//                                                [kc*32 + (lane>>4)*16 + j]
// so a B fragment (K-chunk kc, N-tile nt) is 2 x global_load_b128 per lane.
// ---------------------------------------------------------------------------
__global__ void prep_weights_kernel(const float* __restrict__ W,
                                    bf16_t* __restrict__ WF) {
  int idx = blockIdx.x * blockDim.x + threadIdx.x;
  if (idx >= kDIM * kDIM) return;
  int j    = idx & 15;
  int lane = (idx >> 4) & 31;
  int nt   = (idx >> 9) & 31;
  int kc   = idx >> 14;
  int o = nt * 16 + (lane & 15);
  int i = kc * 32 + (lane >> 4) * 16 + j;
  WF[idx] = f2bf(W[(size_t)o * kDIM + i]);
}

// ---------------------------------------------------------------------------
// Kernel 1: WMMA GEMM  Y[m,n] = X[m,:] . W^T (+bias), per-mode scatter:
//   MODE_Q  : QU=(y+u), QV=(y+v_bias) bf16, row-major [b,h,s,d] (A-side)
//   MODE_K/P: bf16 in B-fragment order KF/PF[bh][dchunk(2)][stile(64)][lane][16]
//   MODE_V  : bf16 in B-fragment order VF[bh][ktile(32)][dsub(4)][lane][16]
//   MODE_OUT: f32 -> d_out
// grid (M/128, N/128), 256 threads (8 waves); wave w owns rows w*16..+16.
// ---------------------------------------------------------------------------
enum ProjMode { MODE_Q = 0, MODE_K = 1, MODE_P = 2, MODE_V = 3, MODE_OUT = 4 };

__global__ __launch_bounds__(256) void proj_gemm_kernel(
    const float* __restrict__ X, const bf16_t* __restrict__ WF,
    const float* __restrict__ bias, const float* __restrict__ ub,
    const float* __restrict__ vb, bf16_t* __restrict__ outA,
    bf16_t* __restrict__ outB, float* __restrict__ outF, int mode) {
  __shared__ bf16_t Asm[128][32];
  const int tid  = threadIdx.x;
  const int lane = tid & 31, wave = tid >> 5;
  const int hh = lane >> 4, rl = lane & 15;
  const int m0 = blockIdx.x * 128, n0 = blockIdx.y * 128;

  v8f acc[8];
#pragma unroll
  for (int i = 0; i < 8; ++i) acc[i] = zero8();

  const int ar = tid >> 1, ac = (tid & 1) * 16;  // A staging: 2 thr/row x 16
  for (int kt = 0; kt < kDIM; kt += 32) {
    {  // stage A block 128x32 (f32 -> bf16, packed b32 LDS stores)
      const float4* s4 =
          (const float4*)(X + (size_t)(m0 + ar) * kDIM + kt + ac);
      if (kt + 32 < kDIM) __builtin_prefetch((const float*)s4 + 32, 0, 0);
#pragma unroll
      for (int t4 = 0; t4 < 4; ++t4) {
        float4 xv = s4[t4];
        *(unsigned*)&Asm[ar][ac + t4 * 4]     = pack2(xv.x, xv.y);
        *(unsigned*)&Asm[ar][ac + t4 * 4 + 2] = pack2(xv.z, xv.w);
      }
    }
    __syncthreads();
    // A fragment: two ds_load_b128
    const int r = wave * 16 + rl;
    Frag a = load_frag(&Asm[r][hh * 16], &Asm[r][hh * 16 + 8]);
    // B fragments straight from L2-resident fragment-ordered weights
    const int kc = kt >> 5;
    const bf16_t* wrow =
        WF + ((size_t)(kc * (kDIM / 16) + (n0 >> 4)) * 32 + lane) * 16;
#pragma unroll
    for (int ns = 0; ns < 8; ++ns) {
      const bf16_t* bp = wrow + (size_t)ns * 512;
      Frag b = load_frag(bp, bp + 8);
      acc[ns] = wmma_bf16(a.v, b.v, acc[ns]);
    }
    __syncthreads();
  }

  // epilogue: C layout -> M = r8 + 8*hh, N = rl
#pragma unroll
  for (int ns = 0; ns < 8; ++ns) {
#pragma unroll
    for (int r8 = 0; r8 < 8; ++r8) {
      int m = m0 + wave * 16 + r8 + 8 * hh;
      int n = n0 + ns * 16 + rl;
      float val = acc[ns][r8] + (bias ? bias[n] : 0.f);
      if (mode == MODE_OUT) {
        outF[(size_t)m * kDIM + n] = val;
      } else {
        int bi = m >> 10, si = m & (kS - 1);
        int head = n >> 6, d = n & (kHD - 1);
        size_t bh = (size_t)bi * kNH + head;
        if (mode == MODE_Q) {
          size_t idx = (bh * kS + si) * kHD + d;
          outA[idx] = f2bf(val + ub[n]);
          outB[idx] = f2bf(val + vb[n]);
        } else if (mode == MODE_V) {
          // VF[bh][t=k/32][dsub=d/16][lane=((k>>4)&1)*16 + (d&15)][j=k&15]
          size_t idx = (((bh * 32 + (si >> 5)) * 4 + (d >> 4)) * 32 +
                        ((si >> 4) & 1) * 16 + (d & 15)) * 16 + (si & 15);
          outA[idx] = f2bf(val);
        } else {
          // KF/PF[bh][c=d/32][st=s/16][lane=((d>>4)&1)*16 + (s&15)][j=d&15]
          size_t idx = (((bh * 2 + (d >> 5)) * 64 + (si >> 4)) * 32 +
                        ((d >> 4) & 1) * 16 + (si & 15)) * 16 + (d & 15);
          outA[idx] = f2bf(val);
        }
      }
    }
  }
}

// ---------------------------------------------------------------------------
// Kernel 2: fused attention per (b, head, 16-row q-tile).
// ---------------------------------------------------------------------------
__global__ __launch_bounds__(256) void attn_fused_kernel(
    const bf16_t* __restrict__ QU, const bf16_t* __restrict__ QV,
    const bf16_t* __restrict__ KF, const bf16_t* __restrict__ PF,
    const bf16_t* __restrict__ VF, const unsigned char* __restrict__ mask,
    float* __restrict__ attn_out, float* __restrict__ ctx_out) {
  extern __shared__ char smem_raw[];
  float*  Cs     = (float*)smem_raw;        // [16][1024] content scores
  float*  Ps     = Cs + 16 * kS;            // [16][1024] raw pos scores
  float*  Pextra = Ps + 16 * kS;            // [1024] raw pos row q0+16
  float*  redM   = Pextra + kS;             // [16][16]
  float*  redS   = redM + 256;              // [16][16]
  float*  ctxp   = redS + 256;              // [2][16][64]
  bf16_t* As     = (bf16_t*)(ctxp + 2 * 16 * kHD);  // [16][1024] attn bf16

  const int tid = threadIdx.x, lane = tid & 31, wave = tid >> 5;
  const int hh = lane >> 4, rl = lane & 15;
  const int q0 = blockIdx.x * 16;
  const int head = blockIdx.y, bz = blockIdx.z;
  const size_t bh = (size_t)bz * kNH + head;
  const bf16_t* QUb = QU + bh * kS * kHD;
  const bf16_t* QVb = QV + bh * kS * kHD;

  // ---- Phase A: content & raw pos tiles via WMMA --------------------------
  Frag aQU0, aQU1, aQV0, aQV1;  // hdim chunks d=0..31, d=32..63
  {
    const bf16_t* r1 = QUb + (size_t)(q0 + rl) * kHD;
    const bf16_t* r2 = QVb + (size_t)(q0 + rl) * kHD;
    aQU0 = load_frag(r1 + hh * 16,      r1 + 8 + hh * 16);
    aQU1 = load_frag(r1 + 32 + hh * 16, r1 + 40 + hh * 16);
    aQV0 = load_frag(r2 + hh * 16,      r2 + 8 + hh * 16);
    aQV1 = load_frag(r2 + 32 + hh * 16, r2 + 40 + hh * 16);
  }
#pragma unroll
  for (int ns = 0; ns < 8; ++ns) {
    const int st = wave * 8 + ns;        // s-tile index, columns st*16..+16
    const bf16_t* kf0 = KF + (((bh * 2 + 0) * 64 + st) * 32 + lane) * 16;
    const bf16_t* kf1 = KF + (((bh * 2 + 1) * 64 + st) * 32 + lane) * 16;
    const bf16_t* pf0 = PF + (((bh * 2 + 0) * 64 + st) * 32 + lane) * 16;
    const bf16_t* pf1 = PF + (((bh * 2 + 1) * 64 + st) * 32 + lane) * 16;
    v8f accC = zero8(), accP = zero8();
    Frag b;
    b = load_frag(kf0, kf0 + 8);  accC = wmma_bf16(aQU0.v, b.v, accC);
    b = load_frag(kf1, kf1 + 8);  accC = wmma_bf16(aQU1.v, b.v, accC);
    b = load_frag(pf0, pf0 + 8);  accP = wmma_bf16(aQV0.v, b.v, accP);
    b = load_frag(pf1, pf1 + 8);  accP = wmma_bf16(aQV1.v, b.v, accP);
    const int kc0 = st * 16;
#pragma unroll
    for (int r8 = 0; r8 < 8; ++r8) {
      int ql = r8 + 8 * hh;
      Cs[ql * kS + kc0 + rl] = accC[r8];
      Ps[ql * kS + kc0 + rl] = accP[r8];
    }
  }
  {  // extra raw-pos row q0+16 through the fragment layout (b128 loads)
    const int qex = q0 + 16;
    if (qex < kS) {
      const bf16_t* qvrow = QVb + (size_t)qex * kHD;
      for (int jj = 0; jj < 4; ++jj) {
        int kc = tid * 4 + jj;
        float acc = 0.f;
#pragma unroll
        for (int c = 0; c < 2; ++c)
#pragma unroll
          for (int h2 = 0; h2 < 2; ++h2) {
            const bf16_t* pf = PF + (((bh * 2 + c) * 64 + (kc >> 4)) * 32 +
                                     h2 * 16 + (kc & 15)) * 16;
            const bf16_t* qv = qvrow + c * 32 + h2 * 16;
            B8 p0 = load8(pf), p1 = load8(pf + 8);
            B8 v0 = load8(qv), v1 = load8(qv + 8);
#pragma unroll
            for (int j = 0; j < 8; ++j)
              acc += bf2f(v0.e[j]) * bf2f(p0.e[j]) +
                     bf2f(v1.e[j]) * bf2f(p1.e[j]);
          }
        Pextra[kc] = acc;
      }
    } else {
      for (int jj = 0; jj < 4; ++jj) Pextra[tid * 4 + jj] = 0.f;
    }
  }
  __syncthreads();

  // ---- Phase B: shift + mask + scale + softmax ----------------------------
  // rel_shift(X)[q,k] = X[q, S-1-(q-k)] if k<=q ; 0 if k==q+1 ; X[q+1, k-q-2]
  const int qr = tid >> 4, seg = tid & 15;
  const int q_abs = q0 + qr;
  float* Crow = Cs + qr * kS;
  const float* Prow  = Ps + qr * kS;
  const float* Pnext = (qr < 15) ? (Ps + (qr + 1) * kS) : Pextra;
  const size_t mbase = ((size_t)bz * kS + q_abs) * kS;
  const uchar4* mrow = (const uchar4*)(mask + mbase);
  float lmax = -1e30f;
  for (int jj = 0; jj < 16; ++jj) {
    const int k0 = seg * 4 + jj * 64;
    float4 cv = *(const float4*)&Crow[k0];
    uchar4 mv = mrow[k0 >> 2];
    float cc[4] = {cv.x, cv.y, cv.z, cv.w};
    const unsigned char mm[4] = {mv.x, mv.y, mv.z, mv.w};
#pragma unroll
    for (int i2 = 0; i2 < 4; ++i2) {
      const int k = k0 + i2;
      float p;
      if (k <= q_abs)          p = Prow[kS - 1 - (q_abs - k)];
      else if (k == q_abs + 1) p = 0.f;
      else                     p = Pnext[k - q_abs - 2];
      float sc = (cc[i2] + p) * kInvSqrtDim;
      if (mm[i2]) sc = -10000.f;
      cc[i2] = sc;
      lmax = fmaxf(lmax, sc);
    }
    float4 ov; ov.x = cc[0]; ov.y = cc[1]; ov.z = cc[2]; ov.w = cc[3];
    *(float4*)&Crow[k0] = ov;
  }
  redM[qr * 16 + seg] = lmax;
  __syncthreads();
  float rmax = -1e30f;
#pragma unroll
  for (int i = 0; i < 16; ++i) rmax = fmaxf(rmax, redM[qr * 16 + i]);
  float lsum = 0.f;
  for (int jj = 0; jj < 16; ++jj) {
    const int k0 = seg * 4 + jj * 64;
    float4 cv = *(const float4*)&Crow[k0];
    cv.x = __expf(cv.x - rmax); cv.y = __expf(cv.y - rmax);
    cv.z = __expf(cv.z - rmax); cv.w = __expf(cv.w - rmax);
    lsum += cv.x + cv.y + cv.z + cv.w;
    *(float4*)&Crow[k0] = cv;
  }
  redS[qr * 16 + seg] = lsum;
  __syncthreads();
  float rsum = 0.f;
#pragma unroll
  for (int i = 0; i < 16; ++i) rsum += redS[qr * 16 + i];
  const float rinv = 1.f / rsum;
  const size_t abase = (bh * kS + q_abs) * kS;
  for (int jj = 0; jj < 16; ++jj) {
    const int k0 = seg * 4 + jj * 64;
    float4 cv = *(const float4*)&Crow[k0];
    cv.x *= rinv; cv.y *= rinv; cv.z *= rinv; cv.w *= rinv;
    *(float4*)&attn_out[abase + k0] = cv;   // bulk HBM write (roofline limiter)
    uint2 pk;
    pk.x = pack2(cv.x, cv.y);
    pk.y = pack2(cv.z, cv.w);
    *(uint2*)&As[qr * kS + k0] = pk;
  }
  __syncthreads();

  // ---- Phase C: ctx = attn . V  (M=16, K=1024, N=64; K split in halves) ---
  {
    const int ns = wave & 3, half = wave >> 2;
    v8f acc = zero8();
    for (int kk = half * 512; kk < half * 512 + 512; kk += 32) {
      const bf16_t* ap = As + rl * kS + kk + 16 * hh;
      Frag a = load_frag(ap, ap + 8);
      const bf16_t* bp =
          VF + (((bh * 32 + (kk >> 5)) * 4 + ns) * 32 + lane) * 16;
      Frag b = load_frag(bp, bp + 8);
      acc = wmma_bf16(a.v, b.v, acc);
    }
#pragma unroll
    for (int r8 = 0; r8 < 8; ++r8) {
      int ql = r8 + 8 * hh;
      ctxp[(half * 16 + ql) * kHD + ns * 16 + rl] = acc[r8];
    }
  }
  __syncthreads();
  for (int jj = 0; jj < 4; ++jj) {
    int idx = tid * 4 + jj;
    int ql = idx >> 6, d = idx & 63;
    float val = ctxp[ql * kHD + d] + ctxp[(16 + ql) * kHD + d];
    ctx_out[((size_t)bz * kS + q0 + ql) * kDIM + head * kHD + d] = val;
  }
}

// ---------------------------------------------------------------------------
// Host launch
// ---------------------------------------------------------------------------
extern "C" void kernel_launch(void* const* d_in, const int* in_sizes, int n_in,
                              void* d_out, int out_size, void* d_ws,
                              size_t ws_size, hipStream_t stream) {
  (void)in_sizes; (void)n_in; (void)out_size; (void)ws_size;
  const float* q    = (const float*)d_in[0];
  const float* k    = (const float*)d_in[1];
  const float* v    = (const float*)d_in[2];
  const float* pos  = (const float*)d_in[3];
  const unsigned char* mask = (const unsigned char*)d_in[4];  // bool tensor
  const float* Wq   = (const float*)d_in[5];
  const float* bq   = (const float*)d_in[6];
  const float* Wk   = (const float*)d_in[7];
  const float* bk   = (const float*)d_in[8];
  const float* Wv   = (const float*)d_in[9];
  const float* bv   = (const float*)d_in[10];
  const float* Wpos = (const float*)d_in[11];
  const float* Wout = (const float*)d_in[12];
  const float* bout = (const float*)d_in[13];
  const float* u    = (const float*)d_in[14];
  const float* vbias= (const float*)d_in[15];

  char* ws = (char*)d_ws;
  size_t off = 0;
  auto carve = [&](size_t bytes) -> char* {
    char* p = ws + off;
    off += (bytes + 255) & ~(size_t)255;
    return p;
  };
  const size_t wbytes = (size_t)kDIM * kDIM * sizeof(bf16_t);
  const size_t hbytes = (size_t)kB * kNH * kS * kHD * sizeof(bf16_t);
  bf16_t* WqF   = (bf16_t*)carve(wbytes);
  bf16_t* WkF   = (bf16_t*)carve(wbytes);
  bf16_t* WvF   = (bf16_t*)carve(wbytes);
  bf16_t* WposF = (bf16_t*)carve(wbytes);
  bf16_t* WoutF = (bf16_t*)carve(wbytes);
  bf16_t* QUws  = (bf16_t*)carve(hbytes);   // (qh+u)      [b,h,s,d]
  bf16_t* QVws  = (bf16_t*)carve(hbytes);   // (qh+v_bias) [b,h,s,d]
  bf16_t* KFws  = (bf16_t*)carve(hbytes);   // kh  B-fragment order
  bf16_t* PFws  = (bf16_t*)carve(hbytes);   // ph  B-fragment order
  bf16_t* VFws  = (bf16_t*)carve(hbytes);   // vh  B-fragment order
  float*  CTXws = (float*)carve((size_t)kB * kS * kDIM * sizeof(float));

  float* ctx_out  = (float*)d_out;                           // [8,1024,512]
  float* attn_out = (float*)d_out + (size_t)kB * kS * kDIM;  // [8,8,1024,1024]

  dim3 blk(256);
  prep_weights_kernel<<<(kDIM * kDIM) / 256, blk, 0, stream>>>(Wq, WqF);
  prep_weights_kernel<<<(kDIM * kDIM) / 256, blk, 0, stream>>>(Wk, WkF);
  prep_weights_kernel<<<(kDIM * kDIM) / 256, blk, 0, stream>>>(Wv, WvF);
  prep_weights_kernel<<<(kDIM * kDIM) / 256, blk, 0, stream>>>(Wpos, WposF);
  prep_weights_kernel<<<(kDIM * kDIM) / 256, blk, 0, stream>>>(Wout, WoutF);

  dim3 gproj(64, 4);
  proj_gemm_kernel<<<gproj, blk, 0, stream>>>(q, WqF, bq, u, vbias,
                                              QUws, QVws, nullptr, MODE_Q);
  proj_gemm_kernel<<<gproj, blk, 0, stream>>>(k, WkF, bk, nullptr, nullptr,
                                              KFws, nullptr, nullptr, MODE_K);
  proj_gemm_kernel<<<gproj, blk, 0, stream>>>(pos, WposF, nullptr, nullptr,
                                              nullptr, PFws, nullptr, nullptr,
                                              MODE_P);
  proj_gemm_kernel<<<gproj, blk, 0, stream>>>(v, WvF, bv, nullptr, nullptr,
                                              VFws, nullptr, nullptr, MODE_V);

  constexpr size_t kAttnSmem =
      (size_t)(16 * kS + 16 * kS + kS + 256 + 256 + 2 * 16 * kHD) *
          sizeof(float) +
      (size_t)16 * kS * sizeof(bf16_t);  // 178176 bytes (gfx1250: 320KB/WGP)
  hipFuncSetAttribute((const void*)attn_fused_kernel,
                      hipFuncAttributeMaxDynamicSharedMemorySize,
                      (int)kAttnSmem);
  dim3 gattn(kS / 16, kNH, kB);
  attn_fused_kernel<<<gattn, blk, kAttnSmem, stream>>>(
      QUws, QVws, KFws, PFws, VFws, mask, attn_out, CTXws);

  proj_gemm_kernel<<<gproj, blk, 0, stream>>>(CTXws, WoutF, bout, nullptr,
                                              nullptr, nullptr, nullptr,
                                              ctx_out, MODE_OUT);
}
